// Net_377957122204
// MI455X (gfx1250) — compile-verified, matrix operand order checked by reading, not run on hardware
//
#include <hip/hip_runtime.h>
#include <math.h>

#define N_NODES 100000
#define N_EDGES 1600000
#define D_IN 512
#define D_HID 128
#define N_CLASSES 100

typedef __attribute__((ext_vector_type(16))) _Float16 v16h;
typedef __attribute__((ext_vector_type(8)))  float    v8f;

// ---------------------------------------------------------------------------
// degree / norm kernels
// ---------------------------------------------------------------------------
__global__ void __launch_bounds__(256) deg_init_kernel(float* deg) {
    int i = blockIdx.x * 256 + threadIdx.x;
    if (i < N_NODES) deg[i] = 1.0f;              // self loop
}

__global__ void __launch_bounds__(256) deg_count_kernel(const int* __restrict__ ei,
                                                        float* deg) {
    int e = blockIdx.x * 256 + threadIdx.x;
    if (e < N_EDGES) {
        int dst = ei[N_EDGES + e];
        unsafeAtomicAdd(&deg[dst], 1.0f);
    }
}

__global__ void __launch_bounds__(256) dinv_kernel(float* deg_dinv) {
    int i = blockIdx.x * 256 + threadIdx.x;
    if (i < N_NODES) {
        float d = deg_dinv[i];
        deg_dinv[i] = (d > 0.f) ? rsqrtf(d) : 0.f;
    }
}

__global__ void __launch_bounds__(256) norm_kernel(const int* __restrict__ ei,
                                                   const float* __restrict__ dinv,
                                                   float* __restrict__ nrm) {
    int e = blockIdx.x * 256 + threadIdx.x;
    if (e < N_EDGES) {
        int src = ei[e];
        int dst = ei[N_EDGES + e];
        nrm[e] = dinv[src] * dinv[dst];
    }
}

// ---------------------------------------------------------------------------
// GEMM1: h = x @ W1   [100000,512] x [512,128] -> [100000,128]
// one wave = 16-row strip x 128 cols (8 WMMA tiles), K chunks of 32
// ---------------------------------------------------------------------------
__global__ void __launch_bounds__(256) gemm1_wmma(const float* __restrict__ x,
                                                  const float* __restrict__ W1,
                                                  float* __restrict__ h) {
    __shared__ _Float16 Bs[D_HID * 32];          // [col][k], 8 KB
    const int wave  = threadIdx.x >> 5;
    const int lane  = threadIdx.x & 31;
    const int strip = blockIdx.x * 8 + wave;     // 16-row strip id
    const bool active = (strip < (N_NODES / 16));
    const int row  = strip * 16 + (lane & 15);
    const int koff = (lane >> 4) * 8;            // lanes 16-31 hold K+8 halves
    const int bcol = lane & 15;

    v8f acc[8] = {};

    for (int k0 = 0; k0 < D_IN; k0 += 32) {
        // stage W1[k0:k0+32, :] into LDS as f16, [col][k] order
        for (int idx = threadIdx.x; idx < D_HID * 32; idx += 256) {
            int c = idx >> 5;
            int k = idx & 31;
            Bs[idx] = (_Float16)W1[(k0 + k) * D_HID + c];
        }
        __syncthreads();

        if (active) {
            const float* xp = x + (size_t)row * D_IN + k0 + koff;
            __builtin_prefetch(xp + 32, 0, 0);   // next K-chunk (global_prefetch_b8)
            float4 f0 = *reinterpret_cast<const float4*>(xp);
            float4 f1 = *reinterpret_cast<const float4*>(xp + 4);
            float4 f2 = *reinterpret_cast<const float4*>(xp + 16);
            float4 f3 = *reinterpret_cast<const float4*>(xp + 20);
            v16h a;
            a[0]=(_Float16)f0.x; a[1]=(_Float16)f0.y; a[2]=(_Float16)f0.z; a[3]=(_Float16)f0.w;
            a[4]=(_Float16)f1.x; a[5]=(_Float16)f1.y; a[6]=(_Float16)f1.z; a[7]=(_Float16)f1.w;
            a[8]=(_Float16)f2.x; a[9]=(_Float16)f2.y; a[10]=(_Float16)f2.z; a[11]=(_Float16)f2.w;
            a[12]=(_Float16)f3.x; a[13]=(_Float16)f3.y; a[14]=(_Float16)f3.z; a[15]=(_Float16)f3.w;

            #pragma unroll
            for (int t = 0; t < 8; ++t) {
                const _Float16* bp = Bs + (size_t)(t * 16 + bcol) * 32;
                v16h b;
                #pragma unroll
                for (int e = 0; e < 8; ++e) b[e]     = bp[koff + e];
                #pragma unroll
                for (int e = 0; e < 8; ++e) b[8 + e] = bp[16 + koff + e];
                acc[t] = __builtin_amdgcn_wmma_f32_16x16x32_f16(
                    false, a, false, b, (short)0, acc[t], false, false);
            }
        }
        __syncthreads();
    }

    if (active) {
        const int rbase = strip * 16 + (lane >> 4) * 8;
        #pragma unroll
        for (int t = 0; t < 8; ++t) {
            #pragma unroll
            for (int j = 0; j < 8; ++j)
                h[(size_t)(rbase + j) * D_HID + t * 16 + bcol] = acc[t][j];
        }
    }
}

// ---------------------------------------------------------------------------
// agg1 init with bias + self-loop:  agg1[i,c] = b1[c] + h[i,c]*dinv[i]^2
// ---------------------------------------------------------------------------
__global__ void __launch_bounds__(256) self_init1(const float* __restrict__ h,
                                                  const float* __restrict__ b1,
                                                  const float* __restrict__ dinv,
                                                  float* __restrict__ agg) {
    int idx = blockIdx.x * 256 + threadIdx.x;    // N*128 threads, exact
    int i = idx >> 7;
    int c = idx & 127;
    float d = dinv[i];
    agg[idx] = b1[c] + h[idx] * d * d;
}

// ---------------------------------------------------------------------------
// scatter1: one wave per edge, lane handles 4 contiguous cols (float4)
// ---------------------------------------------------------------------------
__global__ void __launch_bounds__(256) scatter1(const int* __restrict__ ei,
                                                const float* __restrict__ nrm,
                                                const float* __restrict__ h,
                                                float* __restrict__ agg) {
    int wave = threadIdx.x >> 5;
    int lane = threadIdx.x & 31;
    int e = blockIdx.x * 8 + wave;
    if (e >= N_EDGES) return;                    // wave-uniform
    int src = ei[e];
    int dst = ei[N_EDGES + e];
    float w = nrm[e];
    float4 v = *reinterpret_cast<const float4*>(h + (size_t)src * D_HID + lane * 4);
    float* out = agg + (size_t)dst * D_HID + lane * 4;
    unsafeAtomicAdd(out + 0, v.x * w);
    unsafeAtomicAdd(out + 1, v.y * w);
    unsafeAtomicAdd(out + 2, v.z * w);
    unsafeAtomicAdd(out + 3, v.w * w);
}

// ---------------------------------------------------------------------------
// GEMM2: pre = relu(agg1) @ W2  [100000,128] x [128,100] -> [100000,100]
// 7 column tiles (padded to 112); ReLU fused into the A-fragment convert
// ---------------------------------------------------------------------------
__global__ void __launch_bounds__(256) gemm2_wmma(const float* __restrict__ agg1,
                                                  const float* __restrict__ W2,
                                                  float* __restrict__ pre) {
    __shared__ _Float16 Bs[112 * 32];            // [col][k], 3584 halves -> 7168 B, 7 KB
    const int wave  = threadIdx.x >> 5;
    const int lane  = threadIdx.x & 31;
    const int strip = blockIdx.x * 8 + wave;
    const bool active = (strip < (N_NODES / 16));
    const int row  = strip * 16 + (lane & 15);
    const int koff = (lane >> 4) * 8;
    const int bcol = lane & 15;

    v8f acc[7] = {};

    for (int k0 = 0; k0 < D_HID; k0 += 32) {
        for (int idx = threadIdx.x; idx < 112 * 32; idx += 256) {
            int c = idx >> 5;
            int k = idx & 31;
            float w = (c < N_CLASSES) ? W2[(k0 + k) * N_CLASSES + c] : 0.f;
            Bs[idx] = (_Float16)w;
        }
        __syncthreads();

        if (active) {
            const float* ap = agg1 + (size_t)row * D_HID + k0 + koff;
            float4 f0 = *reinterpret_cast<const float4*>(ap);
            float4 f1 = *reinterpret_cast<const float4*>(ap + 4);
            float4 f2 = *reinterpret_cast<const float4*>(ap + 16);
            float4 f3 = *reinterpret_cast<const float4*>(ap + 20);
            v16h a;   // fused ReLU on convert
            a[0]=(_Float16)fmaxf(f0.x,0.f); a[1]=(_Float16)fmaxf(f0.y,0.f);
            a[2]=(_Float16)fmaxf(f0.z,0.f); a[3]=(_Float16)fmaxf(f0.w,0.f);
            a[4]=(_Float16)fmaxf(f1.x,0.f); a[5]=(_Float16)fmaxf(f1.y,0.f);
            a[6]=(_Float16)fmaxf(f1.z,0.f); a[7]=(_Float16)fmaxf(f1.w,0.f);
            a[8]=(_Float16)fmaxf(f2.x,0.f); a[9]=(_Float16)fmaxf(f2.y,0.f);
            a[10]=(_Float16)fmaxf(f2.z,0.f); a[11]=(_Float16)fmaxf(f2.w,0.f);
            a[12]=(_Float16)fmaxf(f3.x,0.f); a[13]=(_Float16)fmaxf(f3.y,0.f);
            a[14]=(_Float16)fmaxf(f3.z,0.f); a[15]=(_Float16)fmaxf(f3.w,0.f);

            #pragma unroll
            for (int t = 0; t < 7; ++t) {
                const _Float16* bp = Bs + (size_t)(t * 16 + bcol) * 32;
                v16h b;
                #pragma unroll
                for (int e = 0; e < 8; ++e) b[e]     = bp[koff + e];
                #pragma unroll
                for (int e = 0; e < 8; ++e) b[8 + e] = bp[16 + koff + e];
                acc[t] = __builtin_amdgcn_wmma_f32_16x16x32_f16(
                    false, a, false, b, (short)0, acc[t], false, false);
            }
        }
        __syncthreads();
    }

    if (active) {
        const int rbase = strip * 16 + (lane >> 4) * 8;
        #pragma unroll
        for (int t = 0; t < 7; ++t) {
            int col = t * 16 + bcol;
            if (col < N_CLASSES) {
                #pragma unroll
                for (int j = 0; j < 8; ++j)
                    pre[(size_t)(rbase + j) * N_CLASSES + col] = acc[t][j];
            }
        }
    }
}

// ---------------------------------------------------------------------------
// agg2 init:  agg2[i,c] = b2[c] + pre[i,c]*dinv[i]^2
// ---------------------------------------------------------------------------
__global__ void __launch_bounds__(256) self_init2(const float* __restrict__ pre,
                                                  const float* __restrict__ b2,
                                                  const float* __restrict__ dinv,
                                                  float* __restrict__ agg) {
    int idx = blockIdx.x * 256 + threadIdx.x;
    if (idx >= N_NODES * N_CLASSES) return;
    int i = idx / N_CLASSES;
    int c = idx - i * N_CLASSES;
    float d = dinv[i];
    agg[idx] = b2[c] + pre[idx] * d * d;
}

// ---------------------------------------------------------------------------
// scatter2: one wave per edge, lanes 0-24 handle float4 of the 100 cols
// ---------------------------------------------------------------------------
__global__ void __launch_bounds__(256) scatter2(const int* __restrict__ ei,
                                                const float* __restrict__ nrm,
                                                const float* __restrict__ pre,
                                                float* __restrict__ agg) {
    int wave = threadIdx.x >> 5;
    int lane = threadIdx.x & 31;
    int e = blockIdx.x * 8 + wave;
    if (e >= N_EDGES) return;
    int src = ei[e];
    int dst = ei[N_EDGES + e];
    float w = nrm[e];
    if (lane < 25) {                             // 25*4 = 100 cols
        float4 v = *reinterpret_cast<const float4*>(pre + (size_t)src * N_CLASSES + lane * 4);
        float* out = agg + (size_t)dst * N_CLASSES + lane * 4;
        unsafeAtomicAdd(out + 0, v.x * w);
        unsafeAtomicAdd(out + 1, v.y * w);
        unsafeAtomicAdd(out + 2, v.z * w);
        unsafeAtomicAdd(out + 3, v.w * w);
    }
}

// ---------------------------------------------------------------------------
// log_softmax over 100 classes, one wave per node
// ---------------------------------------------------------------------------
__global__ void __launch_bounds__(256) logsoftmax_kernel(const float* __restrict__ agg,
                                                         float* __restrict__ out) {
    int wave = threadIdx.x >> 5;
    int lane = threadIdx.x & 31;
    int i = blockIdx.x * 8 + wave;
    if (i >= N_NODES) return;
    const float* row = agg + (size_t)i * N_CLASSES;
    float v0 = row[lane];
    float v1 = row[lane + 32];
    float v2 = row[lane + 64];
    float v3 = (lane < 4) ? row[lane + 96] : -3.4e38f;
    float m = fmaxf(fmaxf(v0, v1), fmaxf(v2, v3));
    #pragma unroll
    for (int off = 16; off > 0; off >>= 1) m = fmaxf(m, __shfl_xor(m, off, 32));
    float s = __expf(v0 - m) + __expf(v1 - m) + __expf(v2 - m) +
              ((lane < 4) ? __expf(v3 - m) : 0.f);
    #pragma unroll
    for (int off = 16; off > 0; off >>= 1) s += __shfl_xor(s, off, 32);
    float lse = m + __logf(s);
    float* orow = out + (size_t)i * N_CLASSES;
    orow[lane]      = v0 - lse;
    orow[lane + 32] = v1 - lse;
    orow[lane + 64] = v2 - lse;
    if (lane < 4) orow[lane + 96] = v3 - lse;
}

// ---------------------------------------------------------------------------
// launch
// ---------------------------------------------------------------------------
extern "C" void kernel_launch(void* const* d_in, const int* in_sizes, int n_in,
                              void* d_out, int out_size, void* d_ws, size_t ws_size,
                              hipStream_t stream) {
    const float* x  = (const float*)d_in[0];
    const int*   ei = (const int*)d_in[1];      // [2, E] int32 (JAX default x64-off)
    const float* W1 = (const float*)d_in[2];
    const float* b1 = (const float*)d_in[3];
    const float* W2 = (const float*)d_in[4];
    const float* b2 = (const float*)d_in[5];
    float* out = (float*)d_out;

    float* ws   = (float*)d_ws;
    float* hbuf = ws;                                   // N*128 ; reused for `pre` (N*100)
    float* aggb = hbuf + (size_t)N_NODES * D_HID;       // N*128 ; reused for agg2 (N*100)
    float* dinv = aggb + (size_t)N_NODES * D_HID;       // N
    float* nrm  = dinv + N_NODES;                       // E

    // 1. degree (with self loop) -> dinv
    deg_init_kernel<<<(N_NODES + 255) / 256, 256, 0, stream>>>(dinv);
    deg_count_kernel<<<N_EDGES / 256, 256, 0, stream>>>(ei, dinv);
    dinv_kernel<<<(N_NODES + 255) / 256, 256, 0, stream>>>(dinv);
    norm_kernel<<<N_EDGES / 256, 256, 0, stream>>>(ei, dinv, nrm);

    // 2. layer 1
    const int strips = N_NODES / 16;                    // 6250
    gemm1_wmma<<<(strips + 7) / 8, 256, 0, stream>>>(x, W1, hbuf);
    self_init1<<<(N_NODES * D_HID) / 256, 256, 0, stream>>>(hbuf, b1, dinv, aggb);
    scatter1<<<N_EDGES / 8, 256, 0, stream>>>(ei, nrm, hbuf, aggb);

    // 3. layer 2 (ReLU fused into GEMM2 A-load); pre overwrites hbuf
    gemm2_wmma<<<(strips + 7) / 8, 256, 0, stream>>>(aggb, W2, hbuf);
    self_init2<<<(N_NODES * N_CLASSES + 255) / 256, 256, 0, stream>>>(hbuf, b2, dinv, aggb);
    scatter2<<<N_EDGES / 8, 256, 0, stream>>>(ei, nrm, hbuf, aggb);

    // 4. log_softmax -> d_out
    logsoftmax_kernel<<<(N_NODES + 7) / 8, 256, 0, stream>>>(aggb, out);
}
